// EndpointVectorField_57672820850941
// MI455X (gfx1250) — matrix operand order repaired and checked by reading, not compiled, not run on hardware
//
#include <hip/hip_runtime.h>
#include <hip/hip_bf16.h>

// ---------------------------------------------------------------------------
// Types for CDNA5 WMMA
// ---------------------------------------------------------------------------
typedef __attribute__((ext_vector_type(8)))  float   v8f;
typedef __attribute__((ext_vector_type(2)))  float   v2f;
typedef __attribute__((ext_vector_type(16))) __bf16  v16bf;

__device__ __forceinline__ float sigmoidf_(float x){ return 1.f/(1.f+__expf(-x)); }
__device__ __forceinline__ float siluf_(float x){ return x*sigmoidf_(x); }

__device__ __forceinline__ float wsum32(float x){
  #pragma unroll
  for(int m=16;m>=1;m>>=1) x += __shfl_xor(x,m,32);
  return x;
}
__device__ __forceinline__ float wsum16(float x){
  x += __shfl_xor(x,8,32);
  x += __shfl_xor(x,4,32);
  x += __shfl_xor(x,2,32);
  x += __shfl_xor(x,1,32);
  return x;
}

__device__ __forceinline__ v8f wmma_f32x4(v2f a, v2f b, v8f c){
  return __builtin_amdgcn_wmma_f32_16x16x4_f32(false,a,false,b,(short)0,c,false,false);
}
__device__ __forceinline__ v8f wmma_bf16(v16bf a, v16bf b, v8f c){
  return __builtin_amdgcn_wmma_f32_16x16x32_bf16(false,a,false,b,(short)0,c,false,false);
}

struct GvpW { const float *Wh,*Wu,*fW,*fb,*gW,*gb; };
struct Gvp3 { GvpW g[3]; };

#define MWAVES 4

// ---------------------------------------------------------------------------
// bf16 WMMA helper: out = silu(A(16xK) * W(Kx64) + b); A staged (bf16) in LDS
// rows S[m][0..Kp-1] (Kp multiple of 32, rows of W >= Kreal treated as zero).
// Result fragments returned in out[4] (N-tiles of 16) and restaged (bf16)
// into S[m][0..63] for the next layer / the gate matmul.
// ---------------------------------------------------------------------------
template<int LDK>
__device__ __forceinline__ void feats_mm(__bf16 (*S)[LDK], const float* __restrict__ W,
                                         int Kreal, int Kp, const float* __restrict__ bias,
                                         int lane, v8f out[4])
{
  const int lo = lane & 15, half = lane >> 4;
  #pragma unroll
  for(int nb=0;nb<4;++nb){
    v8f acc = {0.f,0.f,0.f,0.f,0.f,0.f,0.f,0.f};
    for(int kb=0;kb<Kp;kb+=32){
      v16bf a;
      const __bf16* row = S[lo];
      #pragma unroll
      for(int vv=0;vv<8;++vv){
        int k = kb + 2*vv + 8*half + ((vv>=4)?8:0);
        a[2*vv]   = row[k];
        a[2*vv+1] = row[k+1];
      }
      v16bf b;
      #pragma unroll
      for(int vv=0;vv<8;++vv){
        int k = kb + 2*vv + 8*half + ((vv>=4)?8:0);
        float x0 = (k   < Kreal) ? W[(size_t)k*64     + nb*16 + lo] : 0.f;
        float x1 = (k+1 < Kreal) ? W[(size_t)(k+1)*64 + nb*16 + lo] : 0.f;
        b[2*vv]   = (__bf16)x0;
        b[2*vv+1] = (__bf16)x1;
      }
      acc = wmma_bf16(a,b,acc);
    }
    float bb = bias[nb*16+lo];
    #pragma unroll
    for(int r=0;r<8;++r) acc[r] = siluf_(acc[r]+bb);
    out[nb] = acc;
  }
  #pragma unroll
  for(int nb=0;nb<4;++nb)
    #pragma unroll
    for(int r=0;r<8;++r) S[r+8*half][nb*16+lo] = (__bf16)out[nb][r];
}

// gate = act(A(16x64) * gW(64x16) + gb) ; A = staged s_out in S[m][0..63]
template<int LDK>
__device__ __forceinline__ void gate_mm(__bf16 (*S)[LDK], const float* __restrict__ gW,
                                        const float* __restrict__ gb, int lane,
                                        float gv[8], bool identity)
{
  const int lo = lane & 15, half = lane >> 4;
  v8f acc = {0.f,0.f,0.f,0.f,0.f,0.f,0.f,0.f};
  #pragma unroll
  for(int kb=0;kb<64;kb+=32){
    v16bf a;
    const __bf16* row = S[lo];
    #pragma unroll
    for(int vv=0;vv<8;++vv){
      int k = kb + 2*vv + 8*half + ((vv>=4)?8:0);
      a[2*vv] = row[k]; a[2*vv+1] = row[k+1];
    }
    v16bf b;
    #pragma unroll
    for(int vv=0;vv<8;++vv){
      int k = kb + 2*vv + 8*half + ((vv>=4)?8:0);
      b[2*vv]   = (__bf16)gW[k*16+lo];
      b[2*vv+1] = (__bf16)gW[(k+1)*16+lo];
    }
    acc = wmma_bf16(a,b,acc);
  }
  float bb = gb[lo];
  #pragma unroll
  for(int r=0;r<8;++r){ float x = acc[r]+bb; gv[r] = identity ? x : sigmoidf_(x); }
}

// ---------------------------------------------------------------------------
// f32 WMMA 16-channel vector mix over 3 coords:
// Vout[m][n][c] = sum_k Vin[m][chInBase+k][c] * W[(wrow0+k)*ldw + n], k=0..15
// optionally writes sh = ||Vh|| (bf16) into S[m][shOff+n]
// ---------------------------------------------------------------------------
__device__ __forceinline__ void vec_mm16(const float (*Vin)[51], float (*Vout)[51],
                                         const float* __restrict__ W, int ldw, int wrow0,
                                         int chInBase, __bf16 (*S)[256], int shOff, int lane)
{
  const int lo = lane & 15, half = lane >> 4;
  v8f a0 = {0.f,0.f,0.f,0.f,0.f,0.f,0.f,0.f};
  v8f a1 = a0, a2 = a0;
  #pragma unroll
  for(int kb=0;kb<16;kb+=4){
    int k = kb + 2*half;
    v2f b; b.x = W[(wrow0+k)*ldw + lo]; b.y = W[(wrow0+k+1)*ldw + lo];
    const float* r0 = &Vin[lo][(chInBase+k)*3];
    const float* r1 = &Vin[lo][(chInBase+k+1)*3];
    v2f ax;
    ax.x = r0[0]; ax.y = r1[0]; a0 = wmma_f32x4(ax,b,a0);
    ax.x = r0[1]; ax.y = r1[1]; a1 = wmma_f32x4(ax,b,a1);
    ax.x = r0[2]; ax.y = r1[2]; a2 = wmma_f32x4(ax,b,a2);
  }
  #pragma unroll
  for(int r=0;r<8;++r){
    int m = r + 8*half;
    float x0=a0[r], x1=a1[r], x2=a2[r];
    float* o = &Vout[m][lo*3];
    o[0]=x0; o[1]=x1; o[2]=x2;
    if (S) S[m][shOff+lo] = (__bf16)__builtin_sqrtf(x0*x0+x1*x1+x2*x2+1e-8f);
  }
}

// ---------------------------------------------------------------------------
// Message kernel: 16 edges per wave; 3 message GVPs fused; atomic aggregation.
// ---------------------------------------------------------------------------
__global__ __launch_bounds__(128)
void k_message(Gvp3 Mw, const int* __restrict__ src, const int* __restrict__ dst,
               const float* __restrict__ h, const float* __restrict__ v,
               const float* __restrict__ e, const float* __restrict__ xd,
               const float* __restrict__ dr,
               float* __restrict__ s_agg, float* __restrict__ V_agg, int E)
{
  __shared__ __bf16 sS [MWAVES][16][256];
  __shared__ float  sVa[MWAVES][16][51];
  __shared__ float  sVb[MWAVES][16][51];
  __shared__ int    sDst[MWAVES][16];

  const int w    = threadIdx.x >> 5;
  const int lane = threadIdx.x & 31;
  const int lo   = lane & 15, half = lane >> 4;
  const int eBase = (blockIdx.x*MWAVES + w)*16;

  // ---- stage per-edge inputs -------------------------------------------
  {
    int eI = eBase + lo; if (eI >= E) eI = E-1;
    int s0 = src[eI], d0 = dst[eI];
    if (!half) sDst[w][lo] = d0;
    __bf16* row = sS[w][lo];
    const float* hs = h + (size_t)s0*64;
    const float* hd = h + (size_t)d0*64;
    const float* ee = e + (size_t)eI*64;
    for (int k=half; k<64; k+=2){
      row[k]      = (__bf16)hs[k];
      row[64+k]   = (__bf16)hd[k];
      row[128+k]  = (__bf16)ee[k];
    }
    const float* dd = dr + (size_t)eI*16;
    for (int k=half; k<16; k+=2) row[192+k] = (__bf16)dd[k];
    for (int k=225+half; k<256; k+=2) row[k] = (__bf16)0.f;
    float* vr = sVa[w][lo];
    if (!half){
      vr[0]=xd[(size_t)eI*3]; vr[1]=xd[(size_t)eI*3+1]; vr[2]=xd[(size_t)eI*3+2];
    }
    const float* vs = v + (size_t)s0*48;
    for (int ch=half; ch<16; ch+=2){
      vr[(1+ch)*3+0]=vs[ch*3+0]; vr[(1+ch)*3+1]=vs[ch*3+1]; vr[(1+ch)*3+2]=vs[ch*3+2];
    }
  }
  __builtin_amdgcn_wave_barrier();

  // ---- GVP1 vector: Vh1 = Wh1^T V  (17 in-ch -> 17 out-ch) -------------
  {
    const float* Wh = Mw.g[0].Wh;           // 17x17
    v8f a0 = {0.f,0.f,0.f,0.f,0.f,0.f,0.f,0.f};
    v8f a1 = a0, a2 = a0;
    #pragma unroll
    for(int kb=0;kb<16;kb+=4){
      int k = kb + 2*half;
      v2f b; b.x = Wh[(1+k)*17 + lo]; b.y = Wh[(2+k)*17 + lo];
      const float* r0 = &sVa[w][lo][(1+k)*3];
      const float* r1 = &sVa[w][lo][(2+k)*3];
      v2f ax;
      ax.x=r0[0]; ax.y=r1[0]; a0 = wmma_f32x4(ax,b,a0);
      ax.x=r0[1]; ax.y=r1[1]; a1 = wmma_f32x4(ax,b,a1);
      ax.x=r0[2]; ax.y=r1[2]; a2 = wmma_f32x4(ax,b,a2);
    }
    float w0 = Wh[lo];                      // Wh[0][n], rank-1 x_diff term
    #pragma unroll
    for(int r=0;r<8;++r){
      int m = r + 8*half;
      const float* x0 = &sVa[w][m][0];
      a0[r] += x0[0]*w0; a1[r] += x0[1]*w0; a2[r] += x0[2]*w0;
    }
    #pragma unroll
    for(int r=0;r<8;++r){
      int m = r + 8*half;
      float x0=a0[r], x1=a1[r], x2=a2[r];
      float* o = &sVb[w][m][lo*3];
      o[0]=x0; o[1]=x1; o[2]=x2;
      sS[w][m][208+lo] = (__bf16)__builtin_sqrtf(x0*x0+x1*x1+x2*x2+1e-8f);
    }
    if (!half){                             // fringe out-channel h=16
      float f0=0,f1=0,f2=0;
      for(int vv=0;vv<17;++vv){
        float ww = Wh[vv*17+16];
        const float* p3 = &sVa[w][lo][vv*3];
        f0 += p3[0]*ww; f1 += p3[1]*ww; f2 += p3[2]*ww;
      }
      sVb[w][lo][48]=f0; sVb[w][lo][49]=f1; sVb[w][lo][50]=f2;
      sS[w][lo][224] = (__bf16)__builtin_sqrtf(f0*f0+f1*f1+f2*f2+1e-8f);
    }
  }
  // Vu1 (17 -> 16): main 16 via WMMA + fringe h=16 rank-1
  vec_mm16(sVb[w], sVa[w], Mw.g[0].Wu, 16, 0, 0, nullptr, 0, lane);
  {
    const float* Wu = Mw.g[0].Wu;           // 17x16
    float wu16 = Wu[16*16+lo];
    #pragma unroll
    for(int r=0;r<8;++r){
      int m = r + 8*half;
      const float* f = &sVb[w][m][48];
      float* o = &sVa[w][m][lo*3];
      o[0]+=f[0]*wu16; o[1]+=f[1]*wu16; o[2]+=f[2]*wu16;
    }
  }

  v8f so[4]; float gv[8];
  // GVP1 scalar: feats (K=225 padded 256) + gate; gate * Vu1
  feats_mm<256>(sS[w], Mw.g[0].fW, 225, 256, Mw.g[0].fb, lane, so);
  gate_mm<256>(sS[w], Mw.g[0].gW, Mw.g[0].gb, lane, gv, false);
  #pragma unroll
  for(int r=0;r<8;++r){
    int m=r+8*half; float g=gv[r];
    float* o=&sVa[w][m][lo*3]; o[0]*=g; o[1]*=g; o[2]*=g;
  }

  // zero pad region [80..95] once (feats2/feats3 read K=96)
  for(int k=80+half;k<96;k+=2) sS[w][lo][k] = (__bf16)0.f;

  // ---- GVP2 -------------------------------------------------------------
  vec_mm16(sVa[w], sVb[w], Mw.g[1].Wh, 16, 0, 0, sS[w], 64, lane);   // Vh2 + sh
  vec_mm16(sVb[w], sVa[w], Mw.g[1].Wu, 16, 0, 0, nullptr, 0, lane);  // Vu2
  feats_mm<256>(sS[w], Mw.g[1].fW, 80, 96, Mw.g[1].fb, lane, so);
  gate_mm<256>(sS[w], Mw.g[1].gW, Mw.g[1].gb, lane, gv, false);
  #pragma unroll
  for(int r=0;r<8;++r){
    int m=r+8*half; float g=gv[r];
    float* o=&sVa[w][m][lo*3]; o[0]*=g; o[1]*=g; o[2]*=g;
  }

  // ---- GVP3 -------------------------------------------------------------
  vec_mm16(sVa[w], sVb[w], Mw.g[2].Wh, 16, 0, 0, sS[w], 64, lane);
  vec_mm16(sVb[w], sVa[w], Mw.g[2].Wu, 16, 0, 0, nullptr, 0, lane);
  feats_mm<256>(sS[w], Mw.g[2].fW, 80, 96, Mw.g[2].fb, lane, so);
  gate_mm<256>(sS[w], Mw.g[2].gW, Mw.g[2].gb, lane, gv, false);

  // ---- atomic segment-sum into dst --------------------------------------
  #pragma unroll
  for(int r=0;r<8;++r){
    int m = r + 8*half;
    if (eBase + m < E){
      int dn = sDst[w][m];
      #pragma unroll
      for(int nb=0;nb<4;++nb)
        atomicAdd(&s_agg[(size_t)dn*64 + nb*16 + lo], so[nb][r]);
      float g = gv[r];
      const float* vu = &sVa[w][m][lo*3];
      float* vd = &V_agg[(size_t)dn*48 + lo*3];
      atomicAdd(&vd[0], g*vu[0]);
      atomicAdd(&vd[1], g*vu[1]);
      atomicAdd(&vd[2], g*vu[2]);
    }
  }
}

// ---------------------------------------------------------------------------
// Edge update: 208->64 silu, 64->64 silu, residual + LN over fragments.
// ---------------------------------------------------------------------------
__global__ __launch_bounds__(128)
void k_edge_update(const float* __restrict__ l1W, const float* __restrict__ l1b,
                   const float* __restrict__ l2W, const float* __restrict__ l2b,
                   const float* __restrict__ lnb, const float* __restrict__ lng,
                   const int* __restrict__ src, const int* __restrict__ dst,
                   const float* __restrict__ h, float* __restrict__ e,
                   const float* __restrict__ dr, int E)
{
  __shared__ __bf16 sS[MWAVES][16][224];
  __shared__ int    sE[MWAVES][16];

  const int w    = threadIdx.x >> 5;
  const int lane = threadIdx.x & 31;
  const int lo   = lane & 15, half = lane >> 4;
  const int eBase = (blockIdx.x*MWAVES + w)*16;

  {
    int eI = eBase + lo; if (eI >= E) eI = E-1;
    int s0 = src[eI], d0 = dst[eI];
    if (!half) sE[w][lo] = eI;
    __bf16* row = sS[w][lo];
    const float* hs = h + (size_t)s0*64;
    const float* hd = h + (size_t)d0*64;
    const float* ee = e + (size_t)eI*64;
    for (int k=half; k<64; k+=2){
      row[k]     = (__bf16)hs[k];
      row[64+k]  = (__bf16)hd[k];
      row[128+k] = (__bf16)ee[k];
    }
    const float* dd = dr + (size_t)eI*16;
    for (int k=half; k<16; k+=2) row[192+k] = (__bf16)dd[k];
    for (int k=208+half; k<224; k+=2) row[k] = (__bf16)0.f;
  }
  __builtin_amdgcn_wave_barrier();

  v8f o1[4], o2[4];
  feats_mm<224>(sS[w], l1W, 208, 224, l1b, lane, o1);
  feats_mm<224>(sS[w], l2W,  64,  64, l2b, lane, o2);

  // residual + LN(64) over fragments
  v8f x[4];
  #pragma unroll
  for(int nb=0;nb<4;++nb)
    #pragma unroll
    for(int r=0;r<8;++r){
      int m = r + 8*half;
      x[nb][r] = e[(size_t)sE[w][m]*64 + nb*16 + lo] + o2[nb][r];
    }
  float mu[8], rv[8];
  #pragma unroll
  for(int r=0;r<8;++r){
    float s = x[0][r]+x[1][r]+x[2][r]+x[3][r];
    mu[r] = wsum16(s)*(1.f/64.f);
  }
  #pragma unroll
  for(int r=0;r<8;++r){
    float s=0.f;
    #pragma unroll
    for(int nb=0;nb<4;++nb){ float dd=x[nb][r]-mu[r]; s+=dd*dd; }
    rv[r] = rsqrtf(wsum16(s)*(1.f/64.f)+1e-5f);
  }
  #pragma unroll
  for(int nb=0;nb<4;++nb)
    #pragma unroll
    for(int r=0;r<8;++r){
      int m = r + 8*half;
      if (eBase + m < E){
        int col = nb*16 + lo;
        e[(size_t)sE[w][m]*64 + col] = (x[nb][r]-mu[r])*rv[r]*lng[col] + lnb[col];
      }
    }
}

// ---------------------------------------------------------------------------
// Node-side kernels (one wave per node/edge; negligible FLOP share)
// ---------------------------------------------------------------------------
__global__ __launch_bounds__(256)
void k_node_embed(const float* __restrict__ a_t, const float* __restrict__ c_t,
                  const float* __restrict__ t, const int* __restrict__ nbi,
                  const float* __restrict__ W1, const float* __restrict__ b1,
                  const float* __restrict__ W2, const float* __restrict__ b2,
                  const float* __restrict__ lnb, const float* __restrict__ lng,
                  float* __restrict__ h, int N)
{
  __shared__ float buf[8][64];
  const int w=threadIdx.x>>5, lane=threadIdx.x&31;
  const int node=blockIdx.x*8+w;
  if(node>=N) return;
  const int o0=2*lane, o1=o0+1;
  float x[17];
  #pragma unroll
  for(int k=0;k<10;++k) x[k]=a_t[(size_t)node*10+k];
  x[10]=t[nbi[node]];
  #pragma unroll
  for(int k=0;k<6;++k) x[11+k]=c_t[(size_t)node*6+k];
  float y0=b1[o0], y1=b1[o1];
  #pragma unroll
  for(int k=0;k<17;++k){ y0+=x[k]*W1[k*64+o0]; y1+=x[k]*W1[k*64+o1]; }
  y0=siluf_(y0); y1=siluf_(y1);
  buf[w][o0]=y0; buf[w][o1]=y1;
  float z0=b2[o0], z1=b2[o1];
  for(int k=0;k<64;++k){ float s=buf[w][k]; z0+=s*W2[k*64+o0]; z1+=s*W2[k*64+o1]; }
  z0=siluf_(z0); z1=siluf_(z1);
  float mu=wsum32(z0+z1)*(1.f/64.f);
  float d0=z0-mu, d1=z1-mu;
  float rvv=rsqrtf(wsum32(d0*d0+d1*d1)*(1.f/64.f)+1e-5f);
  h[(size_t)node*64+o0]=d0*rvv*lng[o0]+lnb[o0];
  h[(size_t)node*64+o1]=d1*rvv*lng[o1]+lnb[o1];
}

__global__ __launch_bounds__(256)
void k_edge_embed(const float* __restrict__ e_t,
                  const float* __restrict__ W1, const float* __restrict__ b1,
                  const float* __restrict__ W2, const float* __restrict__ b2,
                  const float* __restrict__ lnb, const float* __restrict__ lng,
                  float* __restrict__ e, int E)
{
  __shared__ float buf[8][64];
  const int w=threadIdx.x>>5, lane=threadIdx.x&31;
  const int i=blockIdx.x*8+w;
  if(i>=E) return;
  const int o0=2*lane, o1=o0+1;
  float x[5];
  #pragma unroll
  for(int k=0;k<5;++k) x[k]=e_t[(size_t)i*5+k];
  float y0=b1[o0], y1=b1[o1];
  #pragma unroll
  for(int k=0;k<5;++k){ y0+=x[k]*W1[k*64+o0]; y1+=x[k]*W1[k*64+o1]; }
  y0=siluf_(y0); y1=siluf_(y1);
  buf[w][o0]=y0; buf[w][o1]=y1;
  float z0=b2[o0], z1=b2[o1];
  for(int k=0;k<64;++k){ float s=buf[w][k]; z0+=s*W2[k*64+o0]; z1+=s*W2[k*64+o1]; }
  z0=siluf_(z0); z1=siluf_(z1);
  float mu=wsum32(z0+z1)*(1.f/64.f);
  float d0=z0-mu, d1=z1-mu;
  float rvv=rsqrtf(wsum32(d0*d0+d1*d1)*(1.f/64.f)+1e-5f);
  e[(size_t)i*64+o0]=d0*rvv*lng[o0]+lnb[o0];
  e[(size_t)i*64+o1]=d1*rvv*lng[o1]+lnb[o1];
}

__global__ __launch_bounds__(256)
void k_init(const float* __restrict__ x_t, float* __restrict__ pos,
            float* __restrict__ v, int N)
{
  int i = blockIdx.x*256 + threadIdx.x;
  if (i < N*3)  pos[i] = x_t[i];
  if (i < N*48) v[i] = 0.f;
}

__global__ __launch_bounds__(256)
void k_distances(const float* __restrict__ pos, const int* __restrict__ src,
                 const int* __restrict__ dst, float* __restrict__ xd,
                 float* __restrict__ dr, int E)
{
  int i = blockIdx.x*256 + threadIdx.x;
  if (i >= E) return;
  int s = src[i], d0 = dst[i];
  float dx = pos[(size_t)s*3+0]-pos[(size_t)d0*3+0];
  float dy = pos[(size_t)s*3+1]-pos[(size_t)d0*3+1];
  float dz = pos[(size_t)s*3+2]-pos[(size_t)d0*3+2];
  float dij = __builtin_sqrtf(dx*dx+dy*dy+dz*dz+1e-8f)+1e-8f;
  float inv = 1.f/dij;
  xd[(size_t)i*3+0]=dx*inv; xd[(size_t)i*3+1]=dy*inv; xd[(size_t)i*3+2]=dz*inv;
  const float sigma_inv = 16.f/20.f;      // 1/1.25
  #pragma unroll
  for(int k=0;k<16;++k){
    float mu = (20.f/15.f)*(float)k;      // linspace(0,20,16)
    float tt = (dij-mu)*sigma_inv;
    dr[(size_t)i*16+k] = __expf(-tt*tt);
  }
}

__global__ __launch_bounds__(256)
void k_node_update(Gvp3 U, const float* __restrict__ ln1b, const float* __restrict__ ln1g,
                   const float* __restrict__ ln2b, const float* __restrict__ ln2g,
                   float* __restrict__ h, float* __restrict__ v,
                   const float* __restrict__ s_agg, const float* __restrict__ V_agg, int N)
{
  __shared__ float sb[8][64];
  __shared__ float vb[8][48];
  __shared__ float vt[8][48];
  __shared__ float shb[8][16];
  const int w=threadIdx.x>>5, lane=threadIdx.x&31;
  const int node=blockIdx.x*8+w;
  if(node>=N) return;
  const int o0=2*lane, o1=o0+1;

  float x0 = h[(size_t)node*64+o0] + 0.01f*s_agg[(size_t)node*64+o0];
  float x1 = h[(size_t)node*64+o1] + 0.01f*s_agg[(size_t)node*64+o1];
  float mu = wsum32(x0+x1)*(1.f/64.f);
  float d0 = x0-mu, d1 = x1-mu;
  float rvv = rsqrtf(wsum32(d0*d0+d1*d1)*(1.f/64.f)+1e-5f);
  float sres0 = d0*rvv*ln1g[o0]+ln1b[o0];
  float sres1 = d1*rvv*ln1g[o1]+ln1b[o1];
  sb[w][o0]=sres0; sb[w][o1]=sres1;

  float vres[3]={0.f,0.f,0.f};
  if(lane<16){
    #pragma unroll
    for(int c=0;c<3;++c){
      vres[c] = v[(size_t)node*48+lane*3+c] + 0.01f*V_agg[(size_t)node*48+lane*3+c];
      vb[w][lane*3+c]=vres[c];
    }
  }

  #pragma unroll
  for(int gi=0;gi<3;++gi){
    const GvpW p = U.g[gi];
    float vu[3]={0.f,0.f,0.f};
    if(lane<16){
      float a0=0,a1=0,a2=0;
      for(int k=0;k<16;++k){
        float ww=p.Wh[k*16+lane];
        a0+=vb[w][k*3+0]*ww; a1+=vb[w][k*3+1]*ww; a2+=vb[w][k*3+2]*ww;
      }
      vt[w][lane*3+0]=a0; vt[w][lane*3+1]=a1; vt[w][lane*3+2]=a2;
      shb[w][lane]=__builtin_sqrtf(a0*a0+a1*a1+a2*a2+1e-8f);
      float b0=0,b1=0,b2=0;
      for(int k=0;k<16;++k){
        float ww=p.Wu[k*16+lane];
        b0+=vt[w][k*3+0]*ww; b1+=vt[w][k*3+1]*ww; b2+=vt[w][k*3+2]*ww;
      }
      vu[0]=b0; vu[1]=b1; vu[2]=b2;
    }
    float y0=p.fb[o0], y1=p.fb[o1];
    for(int k=0;k<64;++k){ float s=sb[w][k]; y0+=s*p.fW[k*64+o0]; y1+=s*p.fW[k*64+o1]; }
    for(int k=0;k<16;++k){ float s=shb[w][k]; y0+=s*p.fW[(64+k)*64+o0]; y1+=s*p.fW[(64+k)*64+o1]; }
    y0=siluf_(y0); y1=siluf_(y1);
    sb[w][o0]=y0; sb[w][o1]=y1;
    if(lane<16){
      float g=p.gb[lane];
      for(int k=0;k<64;++k) g+=sb[w][k]*p.gW[k*16+lane];
      g=sigmoidf_(g);
      vb[w][lane*3+0]=g*vu[0]; vb[w][lane*3+1]=g*vu[1]; vb[w][lane*3+2]=g*vu[2];
    }
  }

  float z0=sres0+sb[w][o0], z1=sres1+sb[w][o1];
  mu=wsum32(z0+z1)*(1.f/64.f);
  d0=z0-mu; d1=z1-mu;
  rvv=rsqrtf(wsum32(d0*d0+d1*d1)*(1.f/64.f)+1e-5f);
  h[(size_t)node*64+o0]=d0*rvv*ln2g[o0]+ln2b[o0];
  h[(size_t)node*64+o1]=d1*rvv*ln2g[o1]+ln2b[o1];
  if(lane<16){
    #pragma unroll
    for(int c=0;c<3;++c)
      v[(size_t)node*48+lane*3+c]=vres[c]+vb[w][lane*3+c];
  }
}

__global__ __launch_bounds__(256)
void k_pos_update(Gvp3 P, const float* __restrict__ h, const float* __restrict__ v,
                  float* __restrict__ pos, int N)
{
  __shared__ float sb[8][64];
  __shared__ float vb[8][48];
  __shared__ float vt[8][48];
  __shared__ float shb[8][16];
  const int w=threadIdx.x>>5, lane=threadIdx.x&31;
  const int node=blockIdx.x*8+w;
  if(node>=N) return;
  const int o0=2*lane, o1=o0+1;

  sb[w][o0]=h[(size_t)node*64+o0]; sb[w][o1]=h[(size_t)node*64+o1];
  if(lane<16){
    #pragma unroll
    for(int c=0;c<3;++c) vb[w][lane*3+c]=v[(size_t)node*48+lane*3+c];
  }

  #pragma unroll
  for(int gi=0;gi<2;++gi){
    const GvpW p = P.g[gi];
    float vu[3]={0.f,0.f,0.f};
    if(lane<16){
      float a0=0,a1=0,a2=0;
      for(int k=0;k<16;++k){
        float ww=p.Wh[k*16+lane];
        a0+=vb[w][k*3+0]*ww; a1+=vb[w][k*3+1]*ww; a2+=vb[w][k*3+2]*ww;
      }
      vt[w][lane*3+0]=a0; vt[w][lane*3+1]=a1; vt[w][lane*3+2]=a2;
      shb[w][lane]=__builtin_sqrtf(a0*a0+a1*a1+a2*a2+1e-8f);
      float b0=0,b1=0,b2=0;
      for(int k=0;k<16;++k){
        float ww=p.Wu[k*16+lane];
        b0+=vt[w][k*3+0]*ww; b1+=vt[w][k*3+1]*ww; b2+=vt[w][k*3+2]*ww;
      }
      vu[0]=b0; vu[1]=b1; vu[2]=b2;
    }
    float y0=p.fb[o0], y1=p.fb[o1];
    for(int k=0;k<64;++k){ float s=sb[w][k]; y0+=s*p.fW[k*64+o0]; y1+=s*p.fW[k*64+o1]; }
    for(int k=0;k<16;++k){ float s=shb[w][k]; y0+=s*p.fW[(64+k)*64+o0]; y1+=s*p.fW[(64+k)*64+o1]; }
    y0=siluf_(y0); y1=siluf_(y1);
    sb[w][o0]=y0; sb[w][o1]=y1;
    if(lane<16){
      float g=p.gb[lane];
      for(int k=0;k<64;++k) g+=sb[w][k]*p.gW[k*16+lane];
      g=sigmoidf_(g);
      vb[w][lane*3+0]=g*vu[0]; vb[w][lane*3+1]=g*vu[1]; vb[w][lane*3+2]=g*vu[2];
    }
  }
  // last GVP: v_out = 1 channel, identity gate
  {
    const GvpW p = P.g[2];
    if(lane<16){
      float a0=0,a1=0,a2=0;
      for(int k=0;k<16;++k){
        float ww=p.Wh[k*16+lane];
        a0+=vb[w][k*3+0]*ww; a1+=vb[w][k*3+1]*ww; a2+=vb[w][k*3+2]*ww;
      }
      vt[w][lane*3+0]=a0; vt[w][lane*3+1]=a1; vt[w][lane*3+2]=a2;
      shb[w][lane]=__builtin_sqrtf(a0*a0+a1*a1+a2*a2+1e-8f);
    }
    float y0=p.fb[o0], y1=p.fb[o1];
    for(int k=0;k<64;++k){ float s=sb[w][k]; y0+=s*p.fW[k*64+o0]; y1+=s*p.fW[k*64+o1]; }
    for(int k=0;k<16;++k){ float s=shb[w][k]; y0+=s*p.fW[(64+k)*64+o0]; y1+=s*p.fW[(64+k)*64+o1]; }
    y0=siluf_(y0); y1=siluf_(y1);
    float part = y0*p.gW[o0] + y1*p.gW[o1];
    float g = wsum32(part) + p.gb[0];      // identity activation
    if(lane==0){
      float b0=0,b1=0,b2=0;
      for(int k=0;k<16;++k){
        float ww=p.Wu[k];                   // Wu is 16x1
        b0+=vt[w][k*3+0]*ww; b1+=vt[w][k*3+1]*ww; b2+=vt[w][k*3+2]*ww;
      }
      pos[(size_t)node*3+0]+=g*b0;
      pos[(size_t)node*3+1]+=g*b1;
      pos[(size_t)node*3+2]+=g*b2;
    }
  }
}

__global__ __launch_bounds__(256)
void k_node_head(const float* __restrict__ W1, const float* __restrict__ b1,
                 const float* __restrict__ W2, const float* __restrict__ b2,
                 const float* __restrict__ h, const float* __restrict__ pos,
                 float* __restrict__ out, int N)
{
  __shared__ float buf[8][64];
  const int w=threadIdx.x>>5, lane=threadIdx.x&31;
  const int node=blockIdx.x*8+w;
  if(node>=N) return;
  const int o0=2*lane, o1=o0+1;
  buf[w][o0]=h[(size_t)node*64+o0]; buf[w][o1]=h[(size_t)node*64+o1];
  float y0=b1[o0], y1=b1[o1];
  for(int k=0;k<64;++k){ float s=buf[w][k]; y0+=s*W1[k*64+o0]; y1+=s*W1[k*64+o1]; }
  y0=siluf_(y0); y1=siluf_(y1);
  buf[w][o0]=y0; buf[w][o1]=y1;
  if(lane<16){
    float z=b2[lane];
    for(int k=0;k<64;++k) z+=buf[w][k]*W2[k*16+lane];
    if(lane<10) out[90000  + (size_t)node*10 + lane]      = z;
    else        out[390000 + (size_t)node*6  + (lane-10)] = z;
  }
  if(lane<3) out[(size_t)node*3+lane]=pos[(size_t)node*3+lane];
}

__global__ __launch_bounds__(256)
void k_edge_head(const float* __restrict__ W1, const float* __restrict__ b1,
                 const float* __restrict__ W2, const float* __restrict__ b2,
                 const float* __restrict__ e, float* __restrict__ out, int EH)
{
  __shared__ float buf[8][64];
  const int w=threadIdx.x>>5, lane=threadIdx.x&31;
  const int i=blockIdx.x*8+w;
  if(i>=EH) return;
  const int o0=2*lane, o1=o0+1;
  buf[w][o0]=e[(size_t)i*64+o0]+e[((size_t)i+EH)*64+o0];
  buf[w][o1]=e[(size_t)i*64+o1]+e[((size_t)i+EH)*64+o1];
  float y0=b1[o0], y1=b1[o1];
  for(int k=0;k<64;++k){ float s=buf[w][k]; y0+=s*W1[k*64+o0]; y1+=s*W1[k*64+o1]; }
  y0=siluf_(y0); y1=siluf_(y1);
  buf[w][o0]=y0; buf[w][o1]=y1;
  if(lane<5){
    float z=b2[lane];
    for(int k=0;k<64;++k) z+=buf[w][k]*W2[k*5+lane];
    out[570000 + (size_t)i*5 + lane]=z;
  }
}

// ---------------------------------------------------------------------------
// Host launch
// ---------------------------------------------------------------------------
extern "C" void kernel_launch(void* const* d_in, const int* in_sizes, int n_in,
                              void* d_out, int out_size, void* d_ws, size_t ws_size,
                              hipStream_t stream)
{
  (void)in_sizes; (void)n_in; (void)out_size; (void)ws_size;
  constexpr int N = 30000, EH = 480000, E = 960000;

  const float* a_t = (const float*)d_in[0];
  const float* c_t = (const float*)d_in[1];
  const float* e_t = (const float*)d_in[2];
  const float* x_t = (const float*)d_in[3];
  const float* tt  = (const float*)d_in[4];

  int idx = 5;
  auto F = [&](){ return (const float*)d_in[idx++]; };

  struct ConvP { const float *ln1b,*ln1g,*ln2b,*ln2g; Gvp3 msg, upd; } convs[4];
  for(int c=0;c<4;++c){                    // sorted: ln1, ln2, message, update
    convs[c].ln1b=F(); convs[c].ln1g=F();
    convs[c].ln2b=F(); convs[c].ln2g=F();
    for(int g=0;g<3;++g){ GvpW& q=convs[c].msg.g[g];
      q.Wh=F(); q.Wu=F(); q.fW=F(); q.fb=F(); q.gW=F(); q.gb=F(); }
    for(int g=0;g<3;++g){ GvpW& q=convs[c].upd.g[g];
      q.Wh=F(); q.Wu=F(); q.fW=F(); q.fb=F(); q.gW=F(); q.gb=F(); }
  }
  const float* ee_l1W=F(); const float* ee_l1b=F();
  const float* ee_l2W=F(); const float* ee_l2b=F();
  const float* ee_lnb=F(); const float* ee_lng=F();
  const float* eh_l1W=F(); const float* eh_l1b=F();
  const float* eh_l2W=F(); const float* eh_l2b=F();
  const float* eu_l1W=F(); const float* eu_l1b=F();
  const float* eu_l2W=F(); const float* eu_l2b=F();
  const float* eu_lnb=F(); const float* eu_lng=F();
  const float* nh_l1W=F(); const float* nh_l1b=F();
  const float* nh_l2W=F(); const float* nh_l2b=F();
  Gvp3 posu;
  for(int g=0;g<3;++g){ GvpW& q=posu.g[g];
    q.Wh=F(); q.Wu=F(); q.fW=F(); q.fb=F(); q.gW=F(); q.gb=F(); }
  const float* se_l1W=F(); const float* se_l1b=F();
  const float* se_l2W=F(); const float* se_l2b=F();
  const float* se_lnb=F(); const float* se_lng=F();

  const int* src = (const int*)d_in[idx++];
  const int* dst = (const int*)d_in[idx++];
  const int* nbi = (const int*)d_in[idx++];
  // upper_edge_mask unused (it is [ones(EH); zeros(EH)] by construction)

  float* ws    = (float*)d_ws;
  float* pos   = ws;
  float* h     = pos   + (size_t)N*3;
  float* v     = h     + (size_t)N*64;
  float* e     = v     + (size_t)N*48;
  float* xd    = e     + (size_t)E*64;
  float* dr    = xd    + (size_t)E*3;
  float* s_agg = dr    + (size_t)E*16;
  float* V_agg = s_agg + (size_t)N*64;
  float* outF  = (float*)d_out;

  k_node_embed<<<N/8, 256, 0, stream>>>(a_t,c_t,tt,nbi,
      se_l1W,se_l1b,se_l2W,se_l2b,se_lnb,se_lng, h, N);
  k_edge_embed<<<E/8, 256, 0, stream>>>(e_t,
      ee_l1W,ee_l1b,ee_l2W,ee_l2b,ee_lnb,ee_lng, e, E);
  k_init<<<(N*48+255)/256, 256, 0, stream>>>(x_t, pos, v, N);
  k_distances<<<E/256, 256, 0, stream>>>(pos, src, dst, xd, dr, E);

  for(int ci=0; ci<4; ++ci){
    hipMemsetAsync(s_agg, 0, (size_t)(N*64 + N*48)*sizeof(float), stream);
    k_message<<<E/64, 128, 0, stream>>>(convs[ci].msg, src, dst,
        h, v, e, xd, dr, s_agg, V_agg, E);
    k_node_update<<<N/8, 256, 0, stream>>>(convs[ci].upd,
        convs[ci].ln1b, convs[ci].ln1g, convs[ci].ln2b, convs[ci].ln2g,
        h, v, s_agg, V_agg, N);
    if (ci==1 || ci==3){
      k_pos_update<<<N/8, 256, 0, stream>>>(posu, h, v, pos, N);
      k_distances<<<E/256, 256, 0, stream>>>(pos, src, dst, xd, dr, E);
      k_edge_update<<<E/64, 128, 0, stream>>>(eu_l1W,eu_l1b,eu_l2W,eu_l2b,
          eu_lnb,eu_lng, src, dst, h, e, dr, E);
    }
  }

  k_node_head<<<N/8, 256, 0, stream>>>(nh_l1W,nh_l1b,nh_l2W,nh_l2b, h, pos, outF, N);
  k_edge_head<<<EH/8, 256, 0, stream>>>(eh_l1W,eh_l1b,eh_l2W,eh_l2b, e, outF, EH);
}
